// wolo_attention_5428838662468
// MI455X (gfx1250) — matrix operand order repaired and checked by reading, not compiled, not run on hardware
//
#include <hip/hip_runtime.h>

#define L_SEQ 4096
#define DIMD  1024
#define NTOK  16384   // 4 * 4096 tokens
#define NOUT  90      // K*K + K

typedef __attribute__((ext_vector_type(16))) __bf16 v16bf;
typedef __attribute__((ext_vector_type(8)))  float  v8f;
typedef __attribute__((__vector_size__(16))) int    v4i;

struct Frag32 { uint4 lo, hi; };   // 32 bytes == one v16bf fragment

#define AS1 __attribute__((address_space(1)))
#define AS3 __attribute__((address_space(3)))

#if defined(__has_builtin)
#  if __has_builtin(__builtin_amdgcn_global_load_async_to_lds_b128) && \
      __has_builtin(__builtin_amdgcn_s_wait_asynccnt)
#    define USE_ASYNC_LDS 1
#  endif
#endif
#ifndef USE_ASYNC_LDS
#  define USE_ASYNC_LDS 0
#endif

static __device__ __forceinline__ unsigned int f32_to_bf16_rne(float f) {
  union { float f; unsigned int u; } v; v.f = f;
  unsigned int r = v.u + 0x7FFFu + ((v.u >> 16) & 1u);
  return r >> 16;
}
static __device__ __forceinline__ unsigned int pack_bf16x2(float a, float b) {
  return f32_to_bf16_rne(a) | (f32_to_bf16_rne(b) << 16);
}

// ---------------------------------------------------------------------------
// Kernel 0: convert wo (1024x1024 f32) -> bf16, row-major. 1024 blocks x 256.
// ---------------------------------------------------------------------------
__global__ __launch_bounds__(256) void cvt_wo_bf16(
    const float* __restrict__ wsrc, unsigned short* __restrict__ wdst) {
  const int i = blockIdx.x * 256 + threadIdx.x;      // handles 4 elements
  float4 v = ((const float4*)wsrc)[i];
  uint2 p;
  p.x = pack_bf16x2(v.x, v.y);
  p.y = pack_bf16x2(v.z, v.w);
  ((uint2*)wdst)[i] = p;
}

// ---------------------------------------------------------------------------
// Kernel 1: per-token attention coefficients (fp32 for sin() accuracy).
//   att[t,o] = x[t,:] . w2[o,:] + b2[o]   (o = 0..89)
//   coef[t,k] = sum_j sin(att[t, 9k+j])   k = 0..8
//   coef[t,9] = sum att[t, 81..89]
// ---------------------------------------------------------------------------
__global__ __launch_bounds__(256) void coef_kernel(
    const float* __restrict__ x, const float* __restrict__ w2,
    const float* __restrict__ b2, float* __restrict__ coef) {
  __shared__ __align__(16) float xs[8 * DIMD];
  __shared__ float att_s[8 * NOUT];
  const int tid = threadIdx.x;
  const size_t t0 = (size_t)blockIdx.x * 8;

  const float4* xg = (const float4*)(x + t0 * DIMD);
  float4* xs4 = (float4*)xs;
  #pragma unroll
  for (int j = 0; j < 8; ++j) xs4[j * 256 + tid] = xg[j * 256 + tid];
  __syncthreads();

  #pragma unroll 1
  for (int it = 0; it < 3; ++it) {
    const int idx = it * 256 + tid;
    if (idx < 8 * NOUT) {
      const int token = idx / NOUT;
      const int o = idx - token * NOUT;
      const float4* xr = (const float4*)(xs + token * DIMD);
      const float4* wr = (const float4*)(w2 + (size_t)o * DIMD);
      float4 a = {0.f, 0.f, 0.f, 0.f};
      #pragma unroll 4
      for (int kk = 0; kk < DIMD / 4; ++kk) {
        const float4 xv = xr[kk], wv = wr[kk];
        a.x += xv.x * wv.x; a.y += xv.y * wv.y;
        a.z += xv.z * wv.z; a.w += xv.w * wv.w;
      }
      att_s[token * NOUT + o] = b2[o] + ((a.x + a.y) + (a.z + a.w));
    }
  }
  __syncthreads();

  if (tid < 72) {
    const int token = tid / 9, k = tid - token * 9;
    float s = 0.f;
    #pragma unroll
    for (int j = 0; j < 9; ++j) s += sinf(att_s[token * NOUT + k * 9 + j]);
    coef[(t0 + token) * 10 + k] = s;
  } else if (tid < 80) {
    const int token = tid - 72;
    float s = 0.f;
    #pragma unroll
    for (int k = 81; k < 90; ++k) s += att_s[token * NOUT + k];
    coef[(t0 + token) * 10 + 9] = s;
  }
}

// ---------------------------------------------------------------------------
// Kernel 2: window combine, emits y as bf16 (GEMM-2 A matrix).
//   y[t,d] = coef[t,9] + sum_k coef[t,k] * x[b, l+k-4, d]
// ---------------------------------------------------------------------------
__global__ __launch_bounds__(256) void window_kernel(
    const float* __restrict__ x, const float* __restrict__ coef,
    unsigned short* __restrict__ ybf) {
  const int t = blockIdx.x;
  const int b = t >> 12;                // / L_SEQ
  const int l = t & (L_SEQ - 1);
  const int tid = threadIdx.x;

  float c[10];
  #pragma unroll
  for (int i = 0; i < 10; ++i) c[i] = coef[(size_t)t * 10 + i];

  float4 acc = { c[9], c[9], c[9], c[9] };
  const float4* xb = (const float4*)(x + (size_t)b * L_SEQ * DIMD);
  #pragma unroll
  for (int k = 0; k < 9; ++k) {
    const int ll = l + k - 4;           // uniform branch per block
    if (ll >= 0 && ll < L_SEQ) {
      const float4 v = xb[(size_t)ll * (DIMD / 4) + tid];
      acc.x += c[k] * v.x; acc.y += c[k] * v.y;
      acc.z += c[k] * v.z; acc.w += c[k] * v.w;
    }
  }
  uint2 p;
  p.x = pack_bf16x2(acc.x, acc.y);
  p.y = pack_bf16x2(acc.z, acc.w);
  ((uint2*)(ybf + (size_t)t * DIMD))[tid] = p;
}

// ---------------------------------------------------------------------------
// Kernel 3: out[16384x1024] = y_bf16 @ wo_bf16^T + bo, f32 accumulate.
// Block tile 256(M)x128(N), BK=32; 8 waves as 4Mx2N, each wave 64x64
// (16 v_wmma per K step). LDS double-buffered, one barrier per K step.
// Tile k+1 goes global->LDS via GLOBAL_LOAD_ASYNC_TO_LDS_B128 (ASYNCcnt):
// no staging VGPRs, DMA overlaps the WMMA stream on the other buffer.
// ---------------------------------------------------------------------------
__global__ __launch_bounds__(256) void gemm_bf16_wmma(
    const unsigned short* __restrict__ A,   // ybf   [NTOK][1024] bf16
    const unsigned short* __restrict__ B,   // wo_bf [1024][1024] bf16
    const float* __restrict__ bo,
    float* __restrict__ out) {
  constexpr int LSTR  = 80;
  constexpr int ROWS  = 256 + 128;               // A rows + B rows per buffer
  __shared__ __align__(16) unsigned char lds[2 * ROWS * LSTR];   // 60 KB

  const int tid  = threadIdx.x;
  const int bm   = blockIdx.x >> 3;     // 64 M-tiles of 256
  const int bn   = blockIdx.x & 7;      // 8 N-tiles of 128
  const int lane = tid & 31;
  const int wave = tid >> 5;
  const int wm   = wave >> 1;           // 0..3 -> 64 rows each
  const int wn   = wave & 1;            // 0..1 -> 64 cols each
  const int half = lane >> 4;           // 0/1
  const int r    = lane & 15;

  const unsigned short* aBase = A + (size_t)(bm * 256) * DIMD;
  const unsigned short* bBase = B + (size_t)(bn * 128) * DIMD;

  v8f acc[4][4];
  #pragma unroll
  for (int i = 0; i < 4; ++i)
    #pragma unroll
    for (int j = 0; j < 4; ++j)
      acc[i][j] = (v8f){0.f,0.f,0.f,0.f,0.f,0.f,0.f,0.f};

  // chunks 0..1023 -> A (256 rows x 4x16B), 1024..1535 -> B (128 rows x 4x16B)
#if USE_ASYNC_LDS
  auto loadTile = [&](int k0, int buf) {
    AS3 unsigned char* base =
        (AS3 unsigned char*)(lds + buf * (ROWS * LSTR));
    #pragma unroll
    for (int j = 0; j < 6; ++j) {
      const int c   = tid + 256 * j;
      const int row = (j < 4) ? (c >> 2) : (256 + ((c - 1024) >> 2));
      const unsigned short* src =
          (j < 4) ? (aBase + (size_t)(c >> 2) * DIMD + k0 + (c & 3) * 8)
                  : (bBase + (size_t)((c - 1024) >> 2) * DIMD + k0 + (c & 3) * 8);
      __builtin_amdgcn_global_load_async_to_lds_b128(
          (AS1 v4i*)src,
          (AS3 v4i*)(base + row * LSTR + (c & 3) * 16), 0, 0);
    }
    if (k0 + 32 < DIMD) {
      __builtin_prefetch(aBase + (size_t)(tid >> 2) * DIMD + k0 + 32, 0, 3);
      __builtin_prefetch(bBase + (size_t)(tid >> 2) * DIMD + k0 + 32, 0, 3);
    }
  };
  auto tileWait = [&]() { __builtin_amdgcn_s_wait_asynccnt(0); };
#else
  auto loadTile = [&](int k0, int buf) {
    uint4 st[6];
    #pragma unroll
    for (int j = 0; j < 6; ++j) {
      const int c = tid + 256 * j;
      const unsigned short* src =
          (j < 4) ? (aBase + (size_t)(c >> 2) * DIMD + k0 + (c & 3) * 8)
                  : (bBase + (size_t)((c - 1024) >> 2) * DIMD + k0 + (c & 3) * 8);
      st[j] = *(const uint4*)src;
    }
    unsigned char* base = lds + buf * (ROWS * LSTR);
    #pragma unroll
    for (int j = 0; j < 6; ++j) {     // stash immediately: short live range
      const int c   = tid + 256 * j;
      const int row = (j < 4) ? (c >> 2) : (256 + ((c - 1024) >> 2));
      *(uint4*)(base + row * LSTR + (c & 3) * 16) = st[j];
    }
    if (k0 + 32 < DIMD) {
      __builtin_prefetch(aBase + (size_t)(tid >> 2) * DIMD + k0 + 32, 0, 3);
      __builtin_prefetch(bBase + (size_t)(tid >> 2) * DIMD + k0 + 32, 0, 3);
    }
  };
  auto tileWait = [&]() {};
#endif

  auto compute = [&](int buf) {
    const unsigned char* sA = lds + buf * (ROWS * LSTR);
    const unsigned char* sB = sA + 256 * LSTR;
    v16bf af[4], bfr[4];
    // A frag (16x32): lane<16 holds K{0..7,16..23}, lane>=16 K{8..15,24..31}
    #pragma unroll
    for (int mt = 0; mt < 4; ++mt) {
      const unsigned char* p = sA + (wm * 64 + mt * 16 + r) * LSTR + half * 16;
      Frag32 f; f.lo = *(const uint4*)p; f.hi = *(const uint4*)(p + 32);
      af[mt] = __builtin_bit_cast(v16bf, f);
    }
    // B frag (32x16): lane<16 holds K{0..7,16..23}, lane>=16 K{8..15,24..31}
    #pragma unroll
    for (int nt = 0; nt < 4; ++nt) {
      const unsigned char* p = sB + (wn * 64 + nt * 16 + r) * LSTR + half * 16;
      Frag32 f; f.lo = *(const uint4*)p; f.hi = *(const uint4*)(p + 32);
      bfr[nt] = __builtin_bit_cast(v16bf, f);
    }
    #pragma unroll
    for (int mt = 0; mt < 4; ++mt)
      #pragma unroll
      for (int nt = 0; nt < 4; ++nt)
        acc[mt][nt] = __builtin_amdgcn_wmma_f32_16x16x32_bf16(
            false, af[mt], false, bfr[nt], (short)0, acc[mt][nt], false, false);
  };

  // --- pipelined K loop: 32 steps, one barrier each -----------------------
  loadTile(0, 0);
  tileWait();
  __syncthreads();
  #pragma unroll 1
  for (int k = 0; k < DIMD / 32; ++k) {
    const int nxt = k + 1;
    if (nxt < DIMD / 32) loadTile(nxt * 32, nxt & 1);  // into free buffer
    compute(k & 1);
    tileWait();                                        // drain ASYNCcnt
    __syncthreads();
  }

  // --- epilogue: C/D layout VGPR i -> M = i + 8*half, N = r ---------------
  #pragma unroll
  for (int mt = 0; mt < 4; ++mt) {
    #pragma unroll
    for (int nt = 0; nt < 4; ++nt) {
      const int n = bn * 128 + wn * 64 + nt * 16 + r;
      const float bias = bo[n];
      #pragma unroll
      for (int i = 0; i < 8; ++i) {
        const int m = bm * 256 + wm * 64 + mt * 16 + half * 8 + i;
        out[(size_t)m * DIMD + n] = acc[mt][nt][i] + bias;
      }
    }
  }
}

// ---------------------------------------------------------------------------
extern "C" void kernel_launch(void* const* d_in, const int* in_sizes, int n_in,
                              void* d_out, int out_size, void* d_ws, size_t ws_size,
                              hipStream_t stream) {
  (void)in_sizes; (void)n_in; (void)out_size; (void)ws_size;
  const float* x  = (const float*)d_in[0];
  const float* w2 = (const float*)d_in[1];
  const float* b2 = (const float*)d_in[2];
  const float* wo = (const float*)d_in[3];
  const float* bo = (const float*)d_in[4];
  float* out = (float*)d_out;

  char* ws = (char*)d_ws;
  unsigned short* wo_bf = (unsigned short*)ws;                        // 2 MB
  float*          coef  = (float*)(ws + (size_t)(2u << 20));          // 640 KB
  unsigned short* ybf   = (unsigned short*)(ws + (size_t)(3u << 20)); // 32 MB

  cvt_wo_bf16   <<<DIMD * DIMD / (256 * 4), 256, 0, stream>>>(wo, wo_bf);
  coef_kernel   <<<NTOK / 8,                256, 0, stream>>>(x, w2, b2, coef);
  window_kernel <<<NTOK,                    256, 0, stream>>>(x, coef, ybf);
  gemm_bf16_wmma<<<(NTOK / 256) * (DIMD / 128), 256, 0, stream>>>(ybf, wo_bf, bo, out);
}
